// SparseMoeBlock_43963285242699
// MI455X (gfx1250) — matrix operand order repaired and checked by reading, not compiled, not run on hardware
//
#include <hip/hip_runtime.h>
#include <hip/hip_bf16.h>

#define T_TOK 2048
#define DMODEL 2048
#define FFN    4096
#define NEXP   8

typedef __attribute__((ext_vector_type(16))) __bf16 v16bf;
typedef __attribute__((ext_vector_type(8)))  float  v8f;
typedef int v4i __attribute__((vector_size(16)));

union Frag16 { v16bf v; unsigned short u[16]; uint4 q[2]; };
union Pack8  { unsigned short s[8]; uint4 q; };

static __device__ inline unsigned short f2bf(float f) {
    union { float f; unsigned int u; } v; v.f = f;
    unsigned int r = v.u + 0x7FFFu + ((v.u >> 16) & 1u);   // RNE
    return (unsigned short)(r >> 16);
}

// ---------------------------------------------------------------- async LDS copy (gfx1250 path)
#if __has_builtin(__builtin_amdgcn_global_load_async_to_lds_b128)
#define HAVE_ASYNC_LDS 1
#endif

static __device__ inline void copy_b128_to_lds(const unsigned short* gsrc,
                                               unsigned short* lds) {
#if defined(HAVE_ASYNC_LDS)
    typedef __attribute__((address_space(1))) v4i* gv4i_t;
    typedef __attribute__((address_space(3))) v4i* lv4i_t;
    gv4i_t g = (gv4i_t)(__attribute__((address_space(1))) void*)(void*)gsrc;
    // ISA: generic LDS flat addr low 32 bits == LDS offset -> AS(3) pointer
    lv4i_t l = (lv4i_t)(unsigned int)(unsigned long long)lds;
    __builtin_amdgcn_global_load_async_to_lds_b128(g, l, 0, 0);
#else
    *(uint4*)lds = *(const uint4*)gsrc;
#endif
}

static __device__ inline void wait_async_lds() {
#if defined(HAVE_ASYNC_LDS)
# if __has_builtin(__builtin_amdgcn_s_wait_asynccnt)
    __builtin_amdgcn_s_wait_asynccnt(0);
# else
    asm volatile("s_wait_asynccnt 0" ::: "memory");
# endif
#endif
}

// ---------------------------------------------------------------- zero / cvt
__global__ void zero_kernel(float* __restrict__ y, int n, int* __restrict__ cnt) {
    for (int i = blockIdx.x * blockDim.x + threadIdx.x; i < n;
         i += gridDim.x * blockDim.x) y[i] = 0.0f;
    if (blockIdx.x == 0 && threadIdx.x < NEXP) cnt[threadIdx.x] = 0;
}

__global__ void cvt_kernel(const float* __restrict__ x,
                           unsigned short* __restrict__ xb, int n) {
    for (int i = blockIdx.x * blockDim.x + threadIdx.x; i < n;
         i += gridDim.x * blockDim.x) xb[i] = f2bf(x[i]);
}

// ---------------------------------------------------------------- router
__global__ __launch_bounds__(256) void router_kernel(
    const float* __restrict__ x, const float* __restrict__ Wr,
    int* __restrict__ cnt, int* __restrict__ tok, float* __restrict__ scr) {
    const int wv = threadIdx.x >> 5, lane = threadIdx.x & 31;
    const int t = blockIdx.x * 8 + wv;
    if (t >= T_TOK) return;
    float acc[NEXP];
#pragma unroll
    for (int e = 0; e < NEXP; ++e) acc[e] = 0.0f;
    const float* xr = x + (size_t)t * DMODEL;
    for (int d = lane; d < DMODEL; d += 32) {
        const float xv = xr[d];
        const float* wr = Wr + (size_t)d * NEXP;
#pragma unroll
        for (int e = 0; e < NEXP; ++e) acc[e] += xv * wr[e];
    }
#pragma unroll
    for (int e = 0; e < NEXP; ++e)
#pragma unroll
        for (int off = 16; off > 0; off >>= 1)
            acc[e] += __shfl_xor(acc[e], off, 32);

    if (lane == 0) {
        float m = acc[0];
#pragma unroll
        for (int e = 1; e < NEXP; ++e) m = fmaxf(m, acc[e]);
        float g[NEXP], s = 0.0f;
#pragma unroll
        for (int e = 0; e < NEXP; ++e) { g[e] = __expf(acc[e] - m); s += g[e]; }
        const float inv = 1.0f / s;
#pragma unroll
        for (int e = 0; e < NEXP; ++e) g[e] *= inv;
        // faithful to reference: top_k(-gates) => the two SMALLEST gates
        int i0 = 0;
#pragma unroll
        for (int e = 1; e < NEXP; ++e) if (g[e] < g[i0]) i0 = e;
        int i1 = (i0 == 0) ? 1 : 0;
#pragma unroll
        for (int e = 0; e < NEXP; ++e)
            if (e != i0 && g[e] < g[i1]) i1 = e;
        const float denom = 1.0f / (fabsf(g[i0]) + fabsf(g[i1]));
        const float s0 = g[i0] * denom, s1 = g[i1] * denom;

        int slot0 = atomicAdd(&cnt[i0], 1);
        tok[i0 * T_TOK + slot0] = t;  scr[i0 * T_TOK + slot0] = s0;
        int slot1 = atomicAdd(&cnt[i1], 1);
        tok[i1 * T_TOK + slot1] = t;  scr[i1 * T_TOK + slot1] = s1;
    }
}

// ---------------------------------------------------------------- WMMA helpers
static __device__ inline v8f wmma_bf16(const Frag16& a, const Frag16& b, v8f c) {
    return __builtin_amdgcn_wmma_f32_16x16x32_bf16(
        false, a.v, false, b.v, (short)0, c, false, false);
}

// A fragment from row-major LDS tile [rows][40] (80B stride: 16B aligned, conflict-free).
static __device__ inline Frag16 load_a(const unsigned short (*tile)[40],
                                       int mbase, int lane) {
    Frag16 a;
    const int am = mbase + (lane & 15);
    const int kb = (lane >> 4);            // 0/1 -> uint4 idx 0/1
    const uint4* p = (const uint4*)(&tile[am][0]);
    a.q[0] = p[kb];
    a.q[1] = p[2 + kb];
    return a;
}

// B fragment from TRANSPOSED LDS tile [ncols][40]: same K order as load_a.
static __device__ inline Frag16 load_b(const unsigned short (*tileT)[40],
                                       int nbase, int lane) {
    Frag16 b;
    const int bn = nbase + (lane & 15);
    const int ki = (lane >> 4);            // uint4 idx 0/1
    const uint4* p = (const uint4*)(&tileT[bn][0]);
    b.q[0] = p[ki];
    b.q[1] = p[ki + 2];
    return b;
}

// ---------------------------------------------------------------- FFN1: H = silu(X W1) * (X V1)
// block tile 128 (rows) x 64 (F cols); 8 waves, wave = 16-row stripe x 64 cols
__global__ __launch_bounds__(256) void ffn1_kernel(
    const unsigned short* __restrict__ xb,   // [T, D] bf16
    const float* __restrict__ W1,            // [D, F] this expert
    const float* __restrict__ V1,            // [D, F]
    const int* __restrict__ tok,
    const int* __restrict__ cntp,
    unsigned short* __restrict__ H) {        // [T, F] bf16
    const int count = *cntp;
    const int mbase = blockIdx.x * 128;
    if (mbase >= count) return;
    const int nbase = blockIdx.y * 64;

    __shared__ int stok[128];
    __shared__ __attribute__((aligned(16))) unsigned short sx [128][40];
    __shared__ __attribute__((aligned(16))) unsigned short swT[64][40];
    __shared__ __attribute__((aligned(16))) unsigned short svT[64][40];

    const int tid = threadIdx.x;
    if (tid < 128) {
        const int r = mbase + tid;
        stok[tid] = (r < count) ? tok[r] : 0;
    }
    __syncthreads();

    const int lane = tid & 31;
    const int wv   = tid >> 5;              // 16-row stripe

    v8f ah[4], av[4];
#pragma unroll
    for (int j = 0; j < 4; ++j) { ah[j] = (v8f){0,0,0,0,0,0,0,0}; av[j] = (v8f){0,0,0,0,0,0,0,0}; }

    const int lrx = tid >> 1, lcx = (tid & 1) * 16;  // X: 2x b128 per thread
    const int fcol = tid & 63, kg = (tid >> 6) * 8;  // W: 8 strided f32 per thread

    const size_t xrow = (size_t)stok[lrx] * DMODEL;

    for (int k0 = 0; k0 < DMODEL; k0 += 32) {
        // X tile via async DMA to LDS (bf16, no conversion needed)
        copy_b128_to_lds(xb + xrow + k0 + lcx,     &sx[lrx][lcx]);
        copy_b128_to_lds(xb + xrow + k0 + lcx + 8, &sx[lrx][lcx + 8]);
        // W1/V1 tiles, register convert fp32->bf16, written transposed
        const float* w1p = W1 + (size_t)(k0 + kg) * FFN + nbase + fcol;
        const float* v1p = V1 + (size_t)(k0 + kg) * FFN + nbase + fcol;
        Pack8 pw, pv;
#pragma unroll
        for (int j = 0; j < 8; ++j) {
            pw.s[j] = f2bf(w1p[(size_t)j * FFN]);
            pv.s[j] = f2bf(v1p[(size_t)j * FFN]);
        }
        *(uint4*)&swT[fcol][kg] = pw.q;
        *(uint4*)&svT[fcol][kg] = pv.q;
        if (k0 + 32 < DMODEL) {
            __builtin_prefetch(w1p + (size_t)32 * FFN, 0, 1);
            __builtin_prefetch(v1p + (size_t)32 * FFN, 0, 1);
        }
        wait_async_lds();
        __syncthreads();

        const Frag16 a = load_a(sx, wv * 16, lane);
#pragma unroll
        for (int j = 0; j < 4; ++j) {
            const Frag16 bw = load_b(swT, j * 16, lane);
            ah[j] = wmma_bf16(a, bw, ah[j]);
        }
#pragma unroll
        for (int j = 0; j < 4; ++j) {
            const Frag16 bv = load_b(svT, j * 16, lane);
            av[j] = wmma_bf16(a, bv, av[j]);
        }
        __syncthreads();
    }

    // epilogue: silu(h) * v -> H (bf16)
    const int hi = (lane >> 4) * 8;
    const int nlo = lane & 15;
#pragma unroll
    for (int r = 0; r < 8; ++r) {
        const int gr = mbase + wv * 16 + hi + r;
        if (gr < count) {
            unsigned short* hp = H + (size_t)gr * FFN + nbase + nlo;
#pragma unroll
            for (int j = 0; j < 4; ++j) {
                const float h = ah[j][r];
                const float g = (h / (1.0f + __expf(-h))) * av[j][r];
                hp[j * 16] = f2bf(g);
            }
        }
    }
}

// ---------------------------------------------------------------- FFN2: y[tok] += score * (H W2)
__global__ __launch_bounds__(256) void ffn2_kernel(
    const unsigned short* __restrict__ H,    // [T, F] bf16
    const float* __restrict__ W2,            // [F, D] this expert
    const int* __restrict__ tok,
    const float* __restrict__ scr,
    const int* __restrict__ cntp,
    float* __restrict__ y) {                 // [T, D] fp32
    const int count = *cntp;
    const int mbase = blockIdx.x * 128;
    if (mbase >= count) return;
    const int nbase = blockIdx.y * 64;

    __shared__ int   stok[128];
    __shared__ float sscr[128];
    __shared__ __attribute__((aligned(16))) unsigned short sa [128][40];
    __shared__ __attribute__((aligned(16))) unsigned short sbT[64][40];

    const int tid = threadIdx.x;
    if (tid < 128) {
        const int r = mbase + tid;
        stok[tid] = (r < count) ? tok[r] : 0;
        sscr[tid] = (r < count) ? scr[r] : 0.0f;
    }
    __syncthreads();

    const int lane = tid & 31;
    const int wv   = tid >> 5;

    v8f acc[4];
#pragma unroll
    for (int j = 0; j < 4; ++j) acc[j] = (v8f){0,0,0,0,0,0,0,0};

    const int lra = tid >> 1, lca = (tid & 1) * 16;
    const int dcol = tid & 63, kg = (tid >> 6) * 8;

    const size_t hrow = (size_t)(mbase + lra) * FFN;

    for (int k0 = 0; k0 < FFN; k0 += 32) {
        copy_b128_to_lds(H + hrow + k0 + lca,     &sa[lra][lca]);
        copy_b128_to_lds(H + hrow + k0 + lca + 8, &sa[lra][lca + 8]);
        const float* w2p = W2 + (size_t)(k0 + kg) * DMODEL + nbase + dcol;
        Pack8 pb;
#pragma unroll
        for (int j = 0; j < 8; ++j) pb.s[j] = f2bf(w2p[(size_t)j * DMODEL]);
        *(uint4*)&sbT[dcol][kg] = pb.q;
        if (k0 + 32 < FFN)
            __builtin_prefetch(w2p + (size_t)32 * DMODEL, 0, 1);
        wait_async_lds();
        __syncthreads();

        const Frag16 a = load_a(sa, wv * 16, lane);
#pragma unroll
        for (int j = 0; j < 4; ++j) {
            const Frag16 b = load_b(sbT, j * 16, lane);
            acc[j] = wmma_bf16(a, b, acc[j]);
        }
        __syncthreads();
    }

    // experts are stream-serialized; each (token, d) touched once per launch -> plain RMW
    const int hi = (lane >> 4) * 8;
    const int nlo = lane & 15;
#pragma unroll
    for (int r = 0; r < 8; ++r) {
        const int gr = mbase + wv * 16 + hi + r;
        if (gr < count) {
            const int   t = stok[gr - mbase];
            const float s = sscr[gr - mbase];
            float* yp = y + (size_t)t * DMODEL + nbase + nlo;
#pragma unroll
            for (int j = 0; j < 4; ++j) yp[j * 16] += s * acc[j][r];
        }
    }
}

// ---------------------------------------------------------------- launcher
extern "C" void kernel_launch(void* const* d_in, const int* in_sizes, int n_in,
                              void* d_out, int out_size, void* d_ws, size_t ws_size,
                              hipStream_t stream) {
    const float* x  = (const float*)d_in[0];   // [2,1024,2048]
    const float* Wr = (const float*)d_in[1];   // [2048,8]
    const float* W1 = (const float*)d_in[2];   // [8,2048,4096]
    const float* V1 = (const float*)d_in[3];   // [8,2048,4096]
    const float* W2 = (const float*)d_in[4];   // [8,4096,2048]
    float* y = (float*)d_out;                  // [2,1024,2048] fp32

    char* ws = (char*)d_ws;
    unsigned short* xb = (unsigned short*)ws;                      //  8 MB  x bf16
    unsigned short* H  = (unsigned short*)(ws + (8u << 20));       // 16 MB  hidden bf16 (reused per expert)
    int*   tok = (int*)  (ws + (24u << 20));                       // 64 KB
    float* scr = (float*)(ws + (24u << 20) + NEXP * T_TOK * 4);    // 64 KB
    int*   cnt = (int*)  (ws + (24u << 20) + 2 * NEXP * T_TOK * 4);// 32 B

    const int n = T_TOK * DMODEL;
    zero_kernel<<<2048, 256, 0, stream>>>(y, n, cnt);
    cvt_kernel<<<2048, 256, 0, stream>>>(x, xb, n);
    router_kernel<<<T_TOK / 8, 256, 0, stream>>>(x, Wr, cnt, tok, scr);

    for (int e = 0; e < NEXP; ++e) {
        const float* W1e = W1 + (size_t)e * DMODEL * FFN;
        const float* V1e = V1 + (size_t)e * DMODEL * FFN;
        const float* W2e = W2 + (size_t)e * FFN * DMODEL;
        ffn1_kernel<<<dim3(T_TOK / 128, FFN / 64), 256, 0, stream>>>(
            xb, W1e, V1e, tok + e * T_TOK, cnt + e, H);
        ffn2_kernel<<<dim3(T_TOK / 128, DMODEL / 64), 256, 0, stream>>>(
            H, W2e, tok + e * T_TOK, scr + e * T_TOK, cnt + e, y);
    }
}
